// GVAE_64089501991492
// MI455X (gfx1250) — matrix operand order repaired and checked by reading, not compiled, not run on hardware
//
#include <hip/hip_runtime.h>
#include <hip/hip_bf16.h>

typedef __attribute__((ext_vector_type(16))) _Float16 v16h;
typedef __attribute__((ext_vector_type(8)))  float    v8f;

#define WMMA16(a, b, c) __builtin_amdgcn_wmma_f32_16x16x32_f16( \
    false, (a), false, (b), (short)0, (c), false, false)

// ---------------------------------------------------------------------------
// Helpers: CDNA5 WMMA fragment layouts (wave32)
//   A (f16, 16x32): lane L -> M=L%16; half t -> K = (t<8 ? t : t+8) + 8*(L/16)
//   B (f16, 32x16): lane L -> N=L%16; half t -> K = (t<8 ? t : t+8) + 8*(L/16)
//   C/D (f32 16x16): lane L -> N=L%16; vgpr r -> M = r + 8*(L/16)
// ---------------------------------------------------------------------------
__device__ __forceinline__ v16h load_a_frag(const _Float16* act, int ldm,
                                            int kbase, int m, int g) {
  v16h a;
#pragma unroll
  for (int t = 0; t < 16; ++t) {
    int k = kbase + (t < 8 ? t : t + 8) + 8 * g;
    a[t] = act[m * ldm + k];
  }
  return a;
}

__device__ __forceinline__ v16h load_b_frag(const _Float16* w, int ldn,
                                            int kbase, int col, int g) {
  v16h b;
#pragma unroll
  for (int t = 0; t < 16; ++t) {
    int k = kbase + (t < 8 ? t : t + 8) + 8 * g;
    b[t] = w[k * ldn + col];
  }
  return b;
}

__device__ __forceinline__ unsigned pcg_hash(unsigned x) {
  x = x * 747796405u + 2891336453u;
  unsigned w = ((x >> ((x >> 28u) + 4u)) ^ x) * 277803737u;
  return (w >> 22u) ^ w;
}

// ---------------------------------------------------------------------------
// Kernel 1: h = relu(edge_attr @ e_w1 + e_b1)  -> f16 [E,16]
// ---------------------------------------------------------------------------
__global__ void __launch_bounds__(256)
edge_hidden_kernel(const float* __restrict__ ea, const float* __restrict__ w1,
                   const float* __restrict__ b1, _Float16* __restrict__ h,
                   int E) {
  __shared__ float sw[128];
  __shared__ float sb[16];
  int tid = threadIdx.x;
  if (tid < 128) sw[tid] = w1[tid];
  if (tid < 16)  sb[tid] = b1[tid];
  __syncthreads();
  int e = blockIdx.x * 256 + tid;
  if (e >= E) return;
  float a[8];
#pragma unroll
  for (int j = 0; j < 8; ++j) a[j] = ea[e * 8 + j];
#pragma unroll
  for (int c = 0; c < 16; ++c) {
    float acc = sb[c];
#pragma unroll
    for (int j = 0; j < 8; ++j) acc += a[j] * sw[j * 16 + c];
    h[e * 16 + c] = (_Float16)fmaxf(acc, 0.0f);
  }
}

// ---------------------------------------------------------------------------
// Kernel 2: NNConv message + scatter-add via WMMA.
//   msg[16 edges, 16] = U[16, 288] @ T2ext[288, 16]   (9 k-slabs of 32)
//   U[e, j*16+i] = h_e[j] * x[src_e][i];  slab 8 carries x @ B2 (e_b2 term)
// One wave per 16-edge tile; grid-stride over tiles; 4 waves / block.
// ---------------------------------------------------------------------------
__global__ void __launch_bounds__(128)
conv_msg_kernel(const float* __restrict__ xcur, const _Float16* __restrict__ h,
                const float* __restrict__ e_w2, const float* __restrict__ e_b2,
                const int* __restrict__ src, const int* __restrict__ dst,
                float* __restrict__ agg, int E) {
  const int lane  = threadIdx.x & 31;
  const int wave  = threadIdx.x >> 5;
  const int g     = lane >> 4;
  const int m     = lane & 15;
  const int gwave = blockIdx.x * 4 + wave;
  const int nwav  = gridDim.x * 4;

  // B fragments for extended T2 (constant across all edges): 9 slabs.
  v16h B[9];
#pragma unroll
  for (int sl = 0; sl < 9; ++sl) {
#pragma unroll
    for (int t = 0; t < 16; ++t) {
      int k = sl * 32 + (t < 8 ? t : t + 8) + 8 * g;
      float v;
      if (sl < 8) {
        int j = k >> 4, i = k & 15;
        v = e_w2[j * 256 + i * 16 + m];
      } else {
        int kp = k - 256;
        v = (kp < 16) ? e_b2[kp * 16 + m] : 0.0f;
      }
      B[sl][t] = (_Float16)v;
    }
  }

  __shared__ _Float16 hs_all[4][16][16];
  __shared__ _Float16 xs_all[4][16][16];
  __shared__ int      db_all[4][16];
  _Float16 (*hs)[16] = hs_all[wave];
  _Float16 (*xs)[16] = xs_all[wave];
  int* dsb = db_all[wave];

  const int ntiles = (E + 15) >> 4;
  for (int tile = gwave; tile < ntiles; tile += nwav) {
    if (lane < 16) {
      int e  = tile * 16 + lane;
      bool ok = (e < E);
      int sv = ok ? src[e] : 0;
      dsb[lane] = ok ? dst[e] : -1;
#pragma unroll
      for (int j = 0; j < 16; ++j)
        hs[lane][j] = ok ? h[e * 16 + j] : (_Float16)0.0f;
#pragma unroll
      for (int i = 0; i < 16; ++i)
        xs[lane][i] = ok ? (_Float16)xcur[sv * 16 + i] : (_Float16)0.0f;
    }
    asm volatile("s_wait_dscnt 0" ::: "memory");

    v8f c = {0.f, 0.f, 0.f, 0.f, 0.f, 0.f, 0.f, 0.f};
#pragma unroll
    for (int sl = 0; sl < 9; ++sl) {
      v16h a;
#pragma unroll
      for (int t = 0; t < 16; ++t) {
        int k = sl * 32 + (t < 8 ? t : t + 8) + 8 * g;
        _Float16 u;
        if (sl < 8) {
          int j = k >> 4, i = k & 15;
          u = hs[m][j] * xs[m][i];
        } else {
          int kp = k - 256;
          u = (kp < 16) ? xs[m][kp] : (_Float16)0.0f;
        }
        a[t] = u;
      }
      c = WMMA16(a, B[sl], c);
    }

#pragma unroll
    for (int r = 0; r < 8; ++r) {
      int row = r + 8 * g;
      int dv  = dsb[row];
      if (dv >= 0) atomicAdd(&agg[dv * 16 + m], c[r]);
    }
    asm volatile("s_wait_dscnt 0" ::: "memory");
  }
}

// ---------------------------------------------------------------------------
// Kernel 3: t = relu(agg + x @ root + bias); accumulate BN partial stats.
// ---------------------------------------------------------------------------
__global__ void __launch_bounds__(256)
node_update_kernel(const float* __restrict__ xcur, const float* __restrict__ agg,
                   const float* __restrict__ root, const float* __restrict__ bias,
                   float* __restrict__ tbuf, float* __restrict__ stats, int N) {
  __shared__ float sroot[256];
  __shared__ float sbias[16];
  int tid = threadIdx.x;
  sroot[tid] = root[tid];
  if (tid < 16) sbias[tid] = bias[tid];
  __syncthreads();

  int n = blockIdx.x * 256 + tid;
  bool act = (n < N);
  float xv[16];
#pragma unroll
  for (int i = 0; i < 16; ++i) xv[i] = act ? xcur[n * 16 + i] : 0.0f;

  float tv[16];
#pragma unroll
  for (int c = 0; c < 16; ++c) {
    float acc = 0.0f;
    if (act) {
      acc = agg[n * 16 + c] + sbias[c];
#pragma unroll
      for (int i = 0; i < 16; ++i) acc += xv[i] * sroot[i * 16 + c];
      acc = fmaxf(acc, 0.0f);
      tbuf[n * 16 + c] = acc;
    }
    tv[c] = acc;
  }

  int lane = tid & 31;
#pragma unroll
  for (int c = 0; c < 16; ++c) {
    float s1 = tv[c], s2 = tv[c] * tv[c];
#pragma unroll
    for (int off = 16; off > 0; off >>= 1) {
      s1 += __shfl_down(s1, off);
      s2 += __shfl_down(s2, off);
    }
    if (lane == 0) {
      atomicAdd(&stats[c], s1);
      atomicAdd(&stats[16 + c], s2);
    }
  }
}

__global__ void bn_finalize_kernel(float* stats, int N) {
  int c = threadIdx.x;
  if (c < 16) {
    float inv = 1.0f / (float)N;
    float mval = stats[c] * inv;
    float vval = stats[16 + c] * inv - mval * mval;
    stats[c]      = mval;
    stats[16 + c] = rsqrtf(vval + 1e-5f);
  }
}

__global__ void __launch_bounds__(256)
bn_apply_kernel(const float* __restrict__ tbuf, const float* __restrict__ stats,
                const float* __restrict__ gam, const float* __restrict__ bet,
                float* __restrict__ xout, int N) {
  int n = blockIdx.x * 256 + threadIdx.x;
  if (n >= N) return;
#pragma unroll
  for (int c = 0; c < 16; ++c)
    xout[n * 16 + c] =
        (tbuf[n * 16 + c] - stats[c]) * stats[16 + c] * gam[c] + bet[c];
}

// ---------------------------------------------------------------------------
// Kernel 4: VAE heads + reparameterization.  z stored f16 for decoder gathers.
// ---------------------------------------------------------------------------
__global__ void __launch_bounds__(256)
vae_kernel(const float* __restrict__ x, const float* __restrict__ muw,
           const float* __restrict__ mub, const float* __restrict__ lvw,
           const float* __restrict__ lvb, _Float16* __restrict__ z, int N) {
  __shared__ float smu[256], slv[256], smub[16], slvb[16];
  int tid = threadIdx.x;
  smu[tid] = muw[tid];
  slv[tid] = lvw[tid];
  if (tid < 16) { smub[tid] = mub[tid]; slvb[tid] = lvb[tid]; }
  __syncthreads();

  int n = blockIdx.x * 256 + tid;
  if (n >= N) return;
  float xv[16];
#pragma unroll
  for (int i = 0; i < 16; ++i) xv[i] = x[n * 16 + i];

  float eps[16];
#pragma unroll
  for (int p = 0; p < 8; ++p) {
    unsigned base = (unsigned)(n * 8 + p);
    unsigned r1 = pcg_hash(base * 2654435761u + 0x9e3779b9u);
    unsigned r2 = pcg_hash(r1 ^ 0x85ebca6bu);
    float u1 = ((float)r1 + 1.0f) * 2.3283064e-10f;
    float u2 = (float)r2 * 2.3283064e-10f;
    float rad = sqrtf(-2.0f * logf(u1));
    eps[2 * p]     = rad * cosf(6.2831853f * u2);
    eps[2 * p + 1] = rad * sinf(6.2831853f * u2);
  }

#pragma unroll
  for (int c = 0; c < 16; ++c) {
    float mu = smub[c], lv = slvb[c];
#pragma unroll
    for (int i = 0; i < 16; ++i) {
      mu += xv[i] * smu[i * 16 + c];
      lv += xv[i] * slv[i * 16 + c];
    }
    lv = fminf(lv, 10.0f);
    z[n * 16 + c] = (_Float16)(mu + eps[c] * expf(0.5f * lv));
  }
}

// ---------------------------------------------------------------------------
// Kernel 5: edge decoder MLP, WMMA chain, one wave per 16-edge tile.
//   [16x32] -> 64 -> 64 -> 64 -> 64 -> 8   (30 WMMAs per tile)
// ---------------------------------------------------------------------------
__global__ void __launch_bounds__(128)
decoder_kernel(const _Float16* __restrict__ z, const int* __restrict__ src,
               const int* __restrict__ dst,
               const float* __restrict__ dw1, const float* __restrict__ db1,
               const float* __restrict__ dw2, const float* __restrict__ db2,
               const float* __restrict__ dw3, const float* __restrict__ db3,
               const float* __restrict__ dw4, const float* __restrict__ db4,
               const float* __restrict__ dw5, const float* __restrict__ db5,
               float* __restrict__ out, int E) {
  __shared__ _Float16 sw1[32 * 64];
  __shared__ _Float16 sw2[64 * 64];
  __shared__ _Float16 sw3[64 * 64];
  __shared__ _Float16 sw4[64 * 64];
  __shared__ _Float16 sw5[64 * 16];
  __shared__ float    sb[272];            // b1..b4 (4x64), b5 padded to 16
  __shared__ _Float16 zf_all[4][16 * 32];
  __shared__ _Float16 actA_all[4][16 * 64];
  __shared__ _Float16 actB_all[4][16 * 64];

  int tid = threadIdx.x;
  for (int i = tid; i < 2048; i += 128) sw1[i] = (_Float16)dw1[i];
  for (int i = tid; i < 4096; i += 128) {
    sw2[i] = (_Float16)dw2[i];
    sw3[i] = (_Float16)dw3[i];
    sw4[i] = (_Float16)dw4[i];
  }
  for (int i = tid; i < 1024; i += 128) {
    int k = i >> 4, o = i & 15;
    sw5[i] = (o < 8) ? (_Float16)dw5[k * 8 + o] : (_Float16)0.0f;
  }
  for (int i = tid; i < 272; i += 128) {
    float v;
    if (i < 64)        v = db1[i];
    else if (i < 128)  v = db2[i - 64];
    else if (i < 192)  v = db3[i - 128];
    else if (i < 256)  v = db4[i - 192];
    else               v = (i - 256 < 8) ? db5[i - 256] : 0.0f;
    sb[i] = v;
  }
  __syncthreads();

  const int lane  = tid & 31;
  const int wave  = tid >> 5;
  const int g     = lane >> 4;
  const int m     = lane & 15;
  const int gwave = blockIdx.x * 4 + wave;
  const int nwav  = gridDim.x * 4;

  _Float16* zf   = zf_all[wave];
  _Float16* actA = actA_all[wave];
  _Float16* actB = actB_all[wave];

  const int ntiles = (E + 15) >> 4;
  for (int tile = gwave; tile < ntiles; tile += nwav) {
    if (lane < 16) {
      int e  = tile * 16 + lane;
      bool ok = (e < E);
      int sv = ok ? src[e] : 0;
      int dv = ok ? dst[e] : 0;
#pragma unroll
      for (int j = 0; j < 16; ++j) {
        zf[lane * 32 + j]      = ok ? z[sv * 16 + j] : (_Float16)0.0f;
        zf[lane * 32 + 16 + j] = ok ? z[dv * 16 + j] : (_Float16)0.0f;
      }
    }
    asm volatile("s_wait_dscnt 0" ::: "memory");

    // ---- layer 1: [16x32] @ [32x64] ----
    {
      v16h a = load_a_frag(zf, 32, 0, m, g);
#pragma unroll
      for (int ns = 0; ns < 4; ++ns) {
        v16h b = load_b_frag(sw1, 64, 0, ns * 16 + m, g);
        v8f c = {0.f, 0.f, 0.f, 0.f, 0.f, 0.f, 0.f, 0.f};
        c = WMMA16(a, b, c);
        int col = ns * 16 + m;
#pragma unroll
        for (int r = 0; r < 8; ++r)
          actA[(r + 8 * g) * 64 + col] =
              (_Float16)fmaxf(c[r] + sb[col], 0.0f);
      }
    }
    asm volatile("s_wait_dscnt 0" ::: "memory");

    // ---- layers 2..4: [16x64] @ [64x64] ----
    _Float16* ping = actA;
    _Float16* pong = actB;
#pragma unroll
    for (int l = 0; l < 3; ++l) {
      const _Float16* W = (l == 0) ? sw2 : (l == 1) ? sw3 : sw4;
#pragma unroll
      for (int ns = 0; ns < 4; ++ns) {
        v8f c = {0.f, 0.f, 0.f, 0.f, 0.f, 0.f, 0.f, 0.f};
#pragma unroll
        for (int ks = 0; ks < 2; ++ks) {
          v16h a = load_a_frag(ping, 64, ks * 32, m, g);
          v16h b = load_b_frag(W, 64, ks * 32, ns * 16 + m, g);
          c = WMMA16(a, b, c);
        }
        int col = ns * 16 + m;
#pragma unroll
        for (int r = 0; r < 8; ++r)
          pong[(r + 8 * g) * 64 + col] =
              (_Float16)fmaxf(c[r] + sb[64 * (l + 1) + col], 0.0f);
      }
      asm volatile("s_wait_dscnt 0" ::: "memory");
      _Float16* tmp = ping; ping = pong; pong = tmp;
    }

    // ---- layer 5: [16x64] @ [64x8(pad16)] ----
    {
      v8f c = {0.f, 0.f, 0.f, 0.f, 0.f, 0.f, 0.f, 0.f};
#pragma unroll
      for (int ks = 0; ks < 2; ++ks) {
        v16h a = load_a_frag(ping, 64, ks * 32, m, g);
        v16h b = load_b_frag(sw5, 16, ks * 32, m, g);
        c = WMMA16(a, b, c);
      }
      if (m < 8) {
#pragma unroll
        for (int r = 0; r < 8; ++r) {
          int e = tile * 16 + r + 8 * g;
          if (e < E) out[e * 8 + m] = c[r] + sb[256 + m];
        }
      }
    }
    asm volatile("s_wait_dscnt 0" ::: "memory");
  }
}

// ---------------------------------------------------------------------------
// Host-side orchestration
// ---------------------------------------------------------------------------
extern "C" void kernel_launch(void* const* d_in, const int* in_sizes, int n_in,
                              void* d_out, int out_size, void* d_ws,
                              size_t ws_size, hipStream_t stream) {
  const float* x     = (const float*)d_in[0];
  const int*   ei    = (const int*)d_in[1];
  const float* ea    = (const float*)d_in[2];
  const float* e_w1  = (const float*)d_in[3];
  const float* e_b1  = (const float*)d_in[4];
  const float* e_w2  = (const float*)d_in[5];
  const float* e_b2  = (const float*)d_in[6];
  const float* roots[4]  = {(const float*)d_in[7],  (const float*)d_in[11],
                            (const float*)d_in[15], (const float*)d_in[19]};
  const float* biases[4] = {(const float*)d_in[8],  (const float*)d_in[12],
                            (const float*)d_in[16], (const float*)d_in[20]};
  const float* bng[4]    = {(const float*)d_in[9],  (const float*)d_in[13],
                            (const float*)d_in[17], (const float*)d_in[21]};
  const float* bnb[4]    = {(const float*)d_in[10], (const float*)d_in[14],
                            (const float*)d_in[18], (const float*)d_in[22]};
  const float* mu_w = (const float*)d_in[23];
  const float* mu_b = (const float*)d_in[24];
  const float* lv_w = (const float*)d_in[25];
  const float* lv_b = (const float*)d_in[26];
  const float* dw1 = (const float*)d_in[27]; const float* db1 = (const float*)d_in[28];
  const float* dw2 = (const float*)d_in[29]; const float* db2 = (const float*)d_in[30];
  const float* dw3 = (const float*)d_in[31]; const float* db3 = (const float*)d_in[32];
  const float* dw4 = (const float*)d_in[33]; const float* db4 = (const float*)d_in[34];
  const float* dw5 = (const float*)d_in[35]; const float* db5 = (const float*)d_in[36];

  const int N = in_sizes[0] / 16;
  const int E = in_sizes[2] / 8;
  const int* src = ei;
  const int* dst = ei + E;

  // Workspace carve-up (256B aligned)
  char* ws = (char*)d_ws;
  size_t off = 0;
  auto carve = [&](size_t bytes) -> void* {
    void* p = ws + off;
    off += (bytes + 255) & ~(size_t)255;
    return p;
  };
  _Float16* hbuf  = (_Float16*)carve((size_t)E * 16 * sizeof(_Float16));
  float*    agg   = (float*)carve((size_t)N * 16 * sizeof(float));
  float*    tbuf  = (float*)carve((size_t)N * 16 * sizeof(float));
  float*    xA    = (float*)carve((size_t)N * 16 * sizeof(float));
  float*    xB    = (float*)carve((size_t)N * 16 * sizeof(float));
  _Float16* zbuf  = (_Float16*)carve((size_t)N * 16 * sizeof(_Float16));
  float*    stats = (float*)carve(32 * sizeof(float));
  (void)ws_size; (void)n_in; (void)out_size;

  const int nblk = (N + 255) / 256;

  // 1) edge hidden features (shared across all 4 convs)
  edge_hidden_kernel<<<(E + 255) / 256, 256, 0, stream>>>(ea, e_w1, e_b1, hbuf, E);

  // 2) four NNConv + BN layers
  const float* xcur = x;
  float* xnext = xA;
  for (int l = 0; l < 4; ++l) {
    hipMemsetAsync(agg, 0, (size_t)N * 16 * sizeof(float), stream);
    hipMemsetAsync(stats, 0, 32 * sizeof(float), stream);
    conv_msg_kernel<<<1024, 128, 0, stream>>>(xcur, hbuf, e_w2, e_b2, src, dst,
                                              agg, E);
    node_update_kernel<<<nblk, 256, 0, stream>>>(xcur, agg, roots[l], biases[l],
                                                 tbuf, stats, N);
    bn_finalize_kernel<<<1, 32, 0, stream>>>(stats, N);
    bn_apply_kernel<<<nblk, 256, 0, stream>>>(tbuf, stats, bng[l], bnb[l],
                                              xnext, N);
    xcur  = xnext;
    xnext = (xnext == xA) ? xB : xA;
  }

  // 3) VAE heads + reparameterization
  vae_kernel<<<nblk, 256, 0, stream>>>(xcur, mu_w, mu_b, lv_w, lv_b, zbuf, N);

  // 4) edge decoder (WMMA chain)
  decoder_kernel<<<1024, 128, 0, stream>>>(zbuf, src, dst, dw1, db1, dw2, db2,
                                           dw3, db3, dw4, db4, dw5, db5,
                                           (float*)d_out, E);
}
